// GCNConv_3l_128h_w_l128_53781580480523
// MI455X (gfx1250) — matrix operand order repaired and checked by reading, not compiled, not run on hardware
//
#include <hip/hip_runtime.h>
#include <hip/hip_bf16.h>
#include <math.h>

typedef __attribute__((ext_vector_type(16))) _Float16 v16h;
typedef __attribute__((ext_vector_type(8)))  float    v8f;

#define GCN_N   100000
#define GCN_E   800000
#define GCN_F   128

__device__ __forceinline__ void atomAddF(float* p, float v) {
    __hip_atomic_fetch_add(p, v, __ATOMIC_RELAXED, __HIP_MEMORY_SCOPE_AGENT);
}

// ---------------- degree / dinv ----------------
__global__ __launch_bounds__(256) void k_init_deg(float* deg, int n) {
    int i = blockIdx.x * 256 + threadIdx.x;
    if (i < n) deg[i] = 1.0f;   // self-loop weight
}

__global__ __launch_bounds__(256) void k_accum_deg(const int* __restrict__ dst,
                                                   const float* __restrict__ w,
                                                   float* deg, int e) {
    int i = blockIdx.x * 256 + threadIdx.x;
    if (i < e) atomAddF(&deg[dst[i]], w[i]);
}

__global__ __launch_bounds__(256) void k_finalize_dinv(float* deg, int n) {
    int i = blockIdx.x * 256 + threadIdx.x;
    if (i < n) {
        float d = deg[i];
        deg[i] = (d > 0.0f) ? rsqrtf(d) : 0.0f;
    }
}

// ---------------- W transpose (128x128) ----------------
__global__ __launch_bounds__(256) void k_transpose128(const float* __restrict__ W,
                                                      float* __restrict__ Wt) {
    int idx = blockIdx.x * 256 + threadIdx.x;   // 16384 total
    int k = idx >> 7, n = idx & 127;
    Wt[n * 128 + k] = W[k * 128 + n];
}

// ---------------- WMMA GEMM: C[N,128] = A[N,128] * W ----------------
// Wt is W transposed (row n holds column n of W, contiguous in k).
// One wave32 computes one 16x16 tile via v_wmma_f32_16x16x32_f16, K in 4 steps.
__global__ __launch_bounds__(256) void k_gemm_wmma(const float* __restrict__ A,
                                                   const float* __restrict__ Wt,
                                                   float* __restrict__ C,
                                                   int nrows) {
    int wave = blockIdx.x * 8 + (threadIdx.x >> 5);
    int tm = wave >> 3;           // row tile  (nrows/16 of them)
    int tn = wave & 7;            // col tile  (128/16 = 8)
    if (tm * 16 >= nrows) return; // wave-uniform: EXEC stays all-ones for WMMA

    int lane = threadIdx.x & 31;
    int half = lane >> 4;         // 0: lanes 0-15, 1: lanes 16-31
    int r    = lane & 15;         // A-row m / B-col n for this lane

    const float* arow = A  + (size_t)(tm * 16 + r) * GCN_F;
    const float* brow = Wt + (size_t)(tn * 16 + r) * GCN_F;

    v8f acc = {};
    #pragma unroll
    for (int k = 0; k < GCN_F; k += 32) {
        // A fragment: K = [k+half*8, +8) and [k+16+half*8, +8)
        const float4* ap0 = (const float4*)(arow + k + half * 8);
        const float4* ap1 = (const float4*)(arow + k + 16 + half * 8);
        float4 a0 = ap0[0], a1 = ap0[1];
        float4 a2 = ap1[0], a3 = ap1[1];
        // B fragment: K = [k+half*8, +8) and [k+16+half*8, +8)
        const float4* bp0 = (const float4*)(brow + k + half * 8);
        const float4* bp1 = (const float4*)(brow + k + 16 + half * 8);
        float4 b0 = bp0[0], b1 = bp0[1], b2 = bp1[0], b3 = bp1[1];

        v16h a, b;
        a[0]=(_Float16)a0.x; a[1]=(_Float16)a0.y; a[2]=(_Float16)a0.z; a[3]=(_Float16)a0.w;
        a[4]=(_Float16)a1.x; a[5]=(_Float16)a1.y; a[6]=(_Float16)a1.z; a[7]=(_Float16)a1.w;
        a[8]=(_Float16)a2.x; a[9]=(_Float16)a2.y; a[10]=(_Float16)a2.z; a[11]=(_Float16)a2.w;
        a[12]=(_Float16)a3.x; a[13]=(_Float16)a3.y; a[14]=(_Float16)a3.z; a[15]=(_Float16)a3.w;
        b[0]=(_Float16)b0.x; b[1]=(_Float16)b0.y; b[2]=(_Float16)b0.z; b[3]=(_Float16)b0.w;
        b[4]=(_Float16)b1.x; b[5]=(_Float16)b1.y; b[6]=(_Float16)b1.z; b[7]=(_Float16)b1.w;
        b[8]=(_Float16)b2.x; b[9]=(_Float16)b2.y; b[10]=(_Float16)b2.z; b[11]=(_Float16)b2.w;
        b[12]=(_Float16)b3.x; b[13]=(_Float16)b3.y; b[14]=(_Float16)b3.z; b[15]=(_Float16)b3.w;

        acc = __builtin_amdgcn_wmma_f32_16x16x32_f16(
            /*neg_a=*/false, a, /*neg_b=*/false, b,
            /*c_mod=*/(short)0, acc, /*reuse_a=*/false, /*reuse_b=*/false);
    }

    // C/D layout: lane holds col n=r; VGPR v holds row m = v + half*8
    float* cbase = C + (size_t)(tm * 16) * GCN_F + tn * 16 + r;
    #pragma unroll
    for (int v = 0; v < 8; ++v) {
        int m = v + half * 8;
        cbase[(size_t)m * GCN_F] = acc[v];
    }
}

// ---------------- out[i] = b + dinv[i]^2 * xw[i]  (self-loop + bias, inits accumulator)
__global__ __launch_bounds__(256) void k_node_init(const float* __restrict__ xw,
                                                   const float* __restrict__ dinv,
                                                   const float* __restrict__ bias,
                                                   float* __restrict__ out, int n) {
    int idx = blockIdx.x * 256 + threadIdx.x;
    if (idx >= n * GCN_F) return;
    int i = idx >> 7, f = idx & 127;
    float di = dinv[i];
    out[idx] = bias[f] + di * di * xw[idx];
}

// ---------------- edge scatter: one wave32 per edge, float4 per lane ----------------
__global__ __launch_bounds__(256) void k_edge_scatter(const int* __restrict__ src,
                                                      const int* __restrict__ dst,
                                                      const float* __restrict__ w,
                                                      const float* __restrict__ dinv,
                                                      const float* __restrict__ xw,
                                                      float* __restrict__ out, int e) {
    int eid = blockIdx.x * 8 + (threadIdx.x >> 5);
    if (eid >= e) return;
    int lane = threadIdx.x & 31;
    int s = src[eid], d = dst[eid];
    float coef = dinv[s] * w[eid] * dinv[d];
    float4 v = ((const float4*)(xw + (size_t)s * GCN_F))[lane];
    float* orow = out + (size_t)d * GCN_F + lane * 4;
    atomAddF(orow + 0, coef * v.x);
    atomAddF(orow + 1, coef * v.y);
    atomAddF(orow + 2, coef * v.z);
    atomAddF(orow + 3, coef * v.w);
}

// ---------------- ELU in place ----------------
__global__ __launch_bounds__(256) void k_elu(float* h, int total) {
    int idx = blockIdx.x * 256 + threadIdx.x;
    if (idx < total) {
        float v = h[idx];
        h[idx] = (v > 0.0f) ? v : (expf(v) - 1.0f);
    }
}

// ---------------- row L2 normalize in place (one wave32 per row) ----------------
__global__ __launch_bounds__(256) void k_l2norm(float* h, int n) {
    int row = blockIdx.x * 8 + (threadIdx.x >> 5);
    if (row >= n) return;
    int lane = threadIdx.x & 31;
    float4 v = ((const float4*)(h + (size_t)row * GCN_F))[lane];
    float ss = v.x * v.x + v.y * v.y + v.z * v.z + v.w * v.w;
    #pragma unroll
    for (int off = 16; off > 0; off >>= 1) ss += __shfl_xor(ss, off, 32);
    float inv = 1.0f / fmaxf(sqrtf(ss), 1e-12f);
    float4 o = { v.x * inv, v.y * inv, v.z * inv, v.w * inv };
    ((float4*)(h + (size_t)row * GCN_F))[lane] = o;
}

extern "C" void kernel_launch(void* const* d_in, const int* in_sizes, int n_in,
                              void* d_out, int out_size, void* d_ws, size_t ws_size,
                              hipStream_t stream) {
    const float* x    = (const float*)d_in[0];
    const int*   src  = (const int*)  d_in[1];
    const int*   dst  = (const int*)  d_in[2];
    const float* wgt  = (const float*)d_in[3];
    const float* W1   = (const float*)d_in[4];
    const float* b1   = (const float*)d_in[5];
    const float* W2   = (const float*)d_in[6];
    const float* b2   = (const float*)d_in[7];
    const float* W3   = (const float*)d_in[8];
    const float* b3   = (const float*)d_in[9];
    float* out = (float*)d_out;

    const int N = GCN_N, E = GCN_E;

    float* ws   = (float*)d_ws;
    float* dinv = ws;                         // N
    float* Wt   = dinv + N;                   // 128*128
    float* xw   = Wt + 128 * 128;             // N*128
    float* hbuf = xw + (size_t)N * GCN_F;     // N*128

    const int nBlkN   = (N + 255) / 256;           // 391
    const int nBlkE   = (E + 255) / 256;           // 3125
    const int nBlkNF  = (N * GCN_F + 255) / 256;   // 50000
    const int nBlkG   = (N / 16);                  // 6250 blocks * 8 waves = all tiles
    const int nBlkEw  = (E + 7) / 8;               // 100000 (8 edges/block)
    const int nBlkRow = (N + 7) / 8;               // 12500 (8 rows/block)

    // normalization coefficients
    k_init_deg     <<<nBlkN, 256, 0, stream>>>(dinv, N);
    k_accum_deg    <<<nBlkE, 256, 0, stream>>>(dst, wgt, dinv, E);
    k_finalize_dinv<<<nBlkN, 256, 0, stream>>>(dinv, N);

    // ---- layer 1: x -> hbuf (ELU) ----
    k_transpose128 <<<64, 256, 0, stream>>>(W1, Wt);
    k_gemm_wmma    <<<nBlkG, 256, 0, stream>>>(x, Wt, xw, N);
    k_node_init    <<<nBlkNF, 256, 0, stream>>>(xw, dinv, b1, hbuf, N);
    k_edge_scatter <<<nBlkEw, 256, 0, stream>>>(src, dst, wgt, dinv, xw, hbuf, E);
    k_elu          <<<nBlkNF, 256, 0, stream>>>(hbuf, N * GCN_F);

    // ---- layer 2: hbuf -> hbuf (ELU) ----
    k_transpose128 <<<64, 256, 0, stream>>>(W2, Wt);
    k_gemm_wmma    <<<nBlkG, 256, 0, stream>>>(hbuf, Wt, xw, N);
    k_node_init    <<<nBlkNF, 256, 0, stream>>>(xw, dinv, b2, hbuf, N);
    k_edge_scatter <<<nBlkEw, 256, 0, stream>>>(src, dst, wgt, dinv, xw, hbuf, E);
    k_elu          <<<nBlkNF, 256, 0, stream>>>(hbuf, N * GCN_F);

    // ---- layer 3: hbuf -> d_out, then row L2 normalize in place ----
    k_transpose128 <<<64, 256, 0, stream>>>(W3, Wt);
    k_gemm_wmma    <<<nBlkG, 256, 0, stream>>>(hbuf, Wt, xw, N);
    k_node_init    <<<nBlkNF, 256, 0, stream>>>(xw, dinv, b3, out, N);
    k_edge_scatter <<<nBlkEw, 256, 0, stream>>>(src, dst, wgt, dinv, xw, out, E);
    k_l2norm       <<<nBlkRow, 256, 0, stream>>>(out, N);
}
